// GATDetector_7172595384608
// MI455X (gfx1250) — compile-verified, matrix operand order checked by reading, not run on hardware
//
#include <hip/hip_runtime.h>
#include <hip/hip_bf16.h>

typedef __attribute__((ext_vector_type(16))) __bf16 v16bf;
typedef __attribute__((ext_vector_type(8)))  float  v8f;

#define NEG_SLOPE 0.2f
#define BN_EPS 1e-5f

// ---------------------------------------------------------------- utilities

__global__ void fill_f32(float* p, size_t cnt, float v) {
    size_t i = (size_t)blockIdx.x * blockDim.x + threadIdx.x;
    size_t stride = (size_t)gridDim.x * blockDim.x;
    for (; i < cnt; i += stride) p[i] = v;
}

__global__ void f32_to_bf16_k(const float* __restrict__ in, __bf16* __restrict__ out, size_t cnt) {
    size_t i = (size_t)blockIdx.x * blockDim.x + threadIdx.x;
    size_t stride = (size_t)gridDim.x * blockDim.x;
    for (; i < cnt; i += stride) out[i] = (__bf16)in[i];
}

// s -> 1/(s + 1e-16) in place (hoists the per-edge divide out of the scatter pass)
__global__ void recip_k(float* __restrict__ s, size_t cnt) {
    size_t i = (size_t)blockIdx.x * blockDim.x + threadIdx.x;
    if (i < cnt) s[i] = 1.0f / (s[i] + 1e-16f);
}

// ------------------------------------------------- WMMA GEMM: H = X @ W^T
// X: [n,128] bf16 row-major.  W: [OUT,128] fp32 row-major (cached in LDS as bf16).
// Hout: [n,OUT] fp32.  Block = 256 threads = 8 waves, each wave owns a 16-row strip.

template<int OUT>
__global__ __launch_bounds__(256)
void gemm_wmma_bf16(const __bf16* __restrict__ X, const float* __restrict__ W,
                    float* __restrict__ Hout, int n) {
    __shared__ __bf16 sW[OUT * 128];
    for (int i = threadIdx.x; i < OUT * 128; i += 256) sW[i] = (__bf16)W[i];
    __syncthreads();

    const int wave = threadIdx.x >> 5;
    const int lane = threadIdx.x & 31;
    const int half = lane >> 4;
    const int row0 = (blockIdx.x * 8 + wave) * 16;
    if (row0 >= n) return;                       // uniform per wave: EXEC stays full

    constexpr int CT = OUT / 16;
    v8f acc[CT] = {};

    int arow = row0 + (lane & 15);
    if (arow >= n) arow = n - 1;                 // clamp ragged tail (safe re-read)
    const __bf16* Xrow = X + (size_t)arow * 128;

#pragma unroll
    for (int kt = 0; kt < 4; ++kt) {             // K = 128 in 4 steps of 32
        if (kt < 3) __builtin_prefetch(Xrow + (kt + 1) * 32, 0, 3);
        // A fragment: 16x32 bf16, ISA layout (VGPR0-3: K=half*8+0..7, VGPR4-7: +16)
        v16bf a;
        const __bf16* ap = Xrow + kt * 32 + half * 8;
#pragma unroll
        for (int i = 0; i < 8; ++i) a[i] = ap[i];
#pragma unroll
        for (int i = 0; i < 8; ++i) a[8 + i] = ap[16 + i];

#pragma unroll
        for (int ct = 0; ct < CT; ++ct) {
            // B fragment: B[k,o] = W[o,k]; lane owns column o
            const int o = ct * 16 + (lane & 15);
            const __bf16* bp = &sW[o * 128 + kt * 32 + half * 8];
            v16bf b;
#pragma unroll
            for (int i = 0; i < 8; ++i) b[i] = bp[i];
#pragma unroll
            for (int i = 0; i < 8; ++i) b[8 + i] = bp[16 + i];

            acc[ct] = __builtin_amdgcn_wmma_f32_16x16x32_bf16(
                false, a, false, b, (short)0, acc[ct], false, false);
        }
    }

    // D layout: lane -> column, VGPR r -> row r + 8*half
#pragma unroll
    for (int ct = 0; ct < CT; ++ct) {
        const int col = ct * 16 + (lane & 15);
#pragma unroll
        for (int r = 0; r < 8; ++r) {
            const int row = row0 + r + 8 * half;
            if (row < n) Hout[(size_t)row * OUT + col] = acc[ct][r];
        }
    }
}

// -------------------------------------------- attention coefficients per node
__global__ void attn_coeffs(const float* __restrict__ h, const float* __restrict__ a_s,
                            const float* __restrict__ a_d, float* __restrict__ als,
                            float* __restrict__ ald, int n, int H) {
    int node = blockIdx.x * blockDim.x + threadIdx.x;
    if (node >= n) return;
    const float* hp = h + (size_t)node * H * 32;
    for (int hh = 0; hh < H; ++hh) {
        float ss = 0.f, dd = 0.f;
#pragma unroll 8
        for (int f = 0; f < 32; ++f) {
            float v = hp[hh * 32 + f];
            ss += v * a_s[hh * 32 + f];
            dd += v * a_d[hh * 32 + f];
        }
        als[node * H + hh] = ss;
        ald[node * H + hh] = dd;
    }
}

// ------------------------------- edge softmax pass 1: score + segment max
// Stores the leaky-relu score per (edge, head) so later passes stream it
// sequentially instead of re-gathering als[src] / ald[dst].
__global__ void edge_score_max_k(const int* __restrict__ src, const int* __restrict__ dst,
                                 int E, int n, const float* __restrict__ als,
                                 const float* __restrict__ ald, float* __restrict__ m,
                                 float* __restrict__ eval, int H) {
    int e = blockIdx.x * blockDim.x + threadIdx.x;
    if (e >= E + n) return;
    int s_ = (e < E) ? src[e] : (e - E);
    int d_ = (e < E) ? dst[e] : (e - E);
    for (int hh = 0; hh < H; ++hh) {
        float v = als[s_ * H + hh] + ald[d_ * H + hh];
        v = (v > 0.f) ? v : NEG_SLOPE * v;
        eval[(size_t)e * H + hh] = v;
        atomicMax(&m[d_ * H + hh], v);
    }
}

// -------------------- edge softmax pass 2: exp(e - m[dst]) + segment sum
// Overwrites eval in place with the exp numerator.
__global__ void edge_exp_sum_k(const int* __restrict__ dst, int E, int n,
                               const float* __restrict__ m, float* __restrict__ ssum,
                               float* __restrict__ eval, int H) {
    int e = blockIdx.x * blockDim.x + threadIdx.x;
    if (e >= E + n) return;
    int d_ = (e < E) ? dst[e] : (e - E);
    for (int hh = 0; hh < H; ++hh) {
        float ex = __expf(eval[(size_t)e * H + hh] - m[d_ * H + hh]);
        eval[(size_t)e * H + hh] = ex;
        atomicAdd(&ssum[d_ * H + hh], ex);
    }
}

// ----------------------------------------- edge scatter: one wave32 per edge
// alpha = eval[e,h] * rs[dst,h]  (rs = reciprocal segment sum, precomputed)
__global__ __launch_bounds__(256)
void edge_aggregate_k(const int* __restrict__ src, const int* __restrict__ dst,
                      int E, int n, const float* __restrict__ rs,
                      const float* __restrict__ eval, const float* __restrict__ hfeat,
                      float* __restrict__ out, int H) {
    int wid = (int)(((size_t)blockIdx.x * blockDim.x + threadIdx.x) >> 5);
    int lane = threadIdx.x & 31;
    if (wid >= E + n) return;
    int s_ = (wid < E) ? src[wid] : (wid - E);
    int d_ = (wid < E) ? dst[wid] : (wid - E);
    const int F = H * 32;
    for (int f = lane; f < F; f += 32) {
        int hh = f >> 5;
        float alpha = eval[(size_t)wid * H + hh] * rs[d_ * H + hh];
        atomicAdd(&out[(size_t)d_ * F + f], hfeat[(size_t)s_ * F + f] * alpha);
    }
}

// ----------------------------------------------------------- BatchNorm stats
__global__ __launch_bounds__(256)
void bn_stats_k(const float* __restrict__ hagg, int n,
                float* __restrict__ sums, float* __restrict__ sumsq) {
    __shared__ float ls[128], lq[128];
    if (threadIdx.x < 128) { ls[threadIdx.x] = 0.f; lq[threadIdx.x] = 0.f; }
    __syncthreads();
    size_t total = (size_t)n * 128;
    size_t stride = (size_t)gridDim.x * blockDim.x;
    for (size_t i = (size_t)blockIdx.x * blockDim.x + threadIdx.x; i < total; i += stride) {
        float v = hagg[i];
        int c = (int)(i & 127);
        atomicAdd(&ls[c], v);
        atomicAdd(&lq[c], v * v);
    }
    __syncthreads();
    if (threadIdx.x < 128) {
        atomicAdd(&sums[threadIdx.x], ls[threadIdx.x]);
        atomicAdd(&sumsq[threadIdx.x], lq[threadIdx.x]);
    }
}

// ------------------------------------ BN + ELU fused with bf16 downconvert
// (layer-1 bias b1 cancels inside BatchNorm: mean shift removes it, var unchanged)
__global__ void bn_elu_bf16_k(const float* __restrict__ agg, const float* __restrict__ sums,
                              const float* __restrict__ sumsq, const float* __restrict__ gamma,
                              const float* __restrict__ beta, __bf16* __restrict__ out, int n) {
    size_t i = (size_t)blockIdx.x * blockDim.x + threadIdx.x;
    if (i >= (size_t)n * 128) return;
    int c = (int)(i & 127);
    float inv_n = 1.0f / (float)n;
    float mu = sums[c] * inv_n;
    float var = sumsq[c] * inv_n - mu * mu;
    float v = (agg[i] - mu) * rsqrtf(var + BN_EPS) * gamma[c] + beta[c];
    v = (v > 0.f) ? v : (__expf(v) - 1.f);
    out[i] = (__bf16)v;
}

// -------------------------------------- epilogue: +b2, ELU, Wc, log_softmax
__global__ void classify_k(const float* __restrict__ agg2, const float* __restrict__ b2,
                           const float* __restrict__ Wc, const float* __restrict__ bc,
                           float* __restrict__ out, int n) {
    int node = blockIdx.x * blockDim.x + threadIdx.x;
    if (node >= n) return;
    const float* p = agg2 + (size_t)node * 32;
    float l0 = bc[0], l1 = bc[1];
#pragma unroll 8
    for (int i = 0; i < 32; ++i) {
        float v = p[i] + b2[i];
        v = (v > 0.f) ? v : (__expf(v) - 1.f);   // ELU
        l0 += v * Wc[i];
        l1 += v * Wc[32 + i];
    }
    float mx = fmaxf(l0, l1);
    float lse = mx + __logf(__expf(l0 - mx) + __expf(l1 - mx));
    out[node * 2 + 0] = l0 - lse;
    out[node * 2 + 1] = l1 - lse;
}

// ---------------------------------------------------------------- launcher

extern "C" void kernel_launch(void* const* d_in, const int* in_sizes, int n_in,
                              void* d_out, int out_size, void* d_ws, size_t ws_size,
                              hipStream_t stream) {
    const float* x   = (const float*)d_in[0];
    const int*   ei  = (const int*)d_in[1];
    const float* W1  = (const float*)d_in[2];
    const float* a1s = (const float*)d_in[3];
    const float* a1d = (const float*)d_in[4];
    // d_in[5] = b1 : cancels inside BatchNorm, intentionally unused
    const float* gamma = (const float*)d_in[6];
    const float* beta  = (const float*)d_in[7];
    const float* W2  = (const float*)d_in[8];
    const float* a2s = (const float*)d_in[9];
    const float* a2d = (const float*)d_in[10];
    const float* b2  = (const float*)d_in[11];
    const float* Wc  = (const float*)d_in[12];
    const float* bc  = (const float*)d_in[13];
    float* out = (float*)d_out;

    const int n = in_sizes[0] / 128;       // nodes
    const int E = in_sizes[1] / 2;         // raw edges
    const int EE = E + n;                  // with self loops
    const int* src = ei;
    const int* dst = ei + E;

    // ---- workspace carve-out (256B aligned)
    size_t off = 0;
    auto carve = [&](size_t bytes) -> void* {
        void* p = (char*)d_ws + off;
        off += (bytes + 255) & ~(size_t)255;
        return p;
    };
    __bf16* x16   = (__bf16*)carve((size_t)n * 128 * 2);   // bf16 features (reused for BN out)
    float*  h1    = (float*) carve((size_t)n * 128 * 4);   // GEMM1 out (reused for GEMM2 out)
    float*  agg1  = (float*) carve((size_t)n * 128 * 4);
    float*  eval  = (float*) carve((size_t)EE * 4 * 4);    // per-edge scores (H=4 worst case)
    float*  als1  = (float*) carve((size_t)n * 4 * 4);
    float*  ald1  = (float*) carve((size_t)n * 4 * 4);
    float*  m1    = (float*) carve((size_t)n * 4 * 4);
    float*  s1    = (float*) carve((size_t)n * 4 * 4);
    float*  sums  = (float*) carve(128 * 4);
    float*  sumsq = (float*) carve(128 * 4);
    float*  als2  = (float*) carve((size_t)n * 4);
    float*  ald2  = (float*) carve((size_t)n * 4);
    float*  m2    = (float*) carve((size_t)n * 4);
    float*  s2    = (float*) carve((size_t)n * 4);
    float*  agg2  = (float*) carve((size_t)n * 32 * 4);
    float*  h2    = h1;                                    // [n,32] overlays h1

    const int B = 256;
    auto grid1 = [&](size_t work) { return (unsigned)((work + B - 1) / B); };

    // ---- init accumulators
    fill_f32<<<grid1((size_t)n * 128), B, 0, stream>>>(agg1, (size_t)n * 128, 0.f);
    fill_f32<<<grid1((size_t)n * 32),  B, 0, stream>>>(agg2, (size_t)n * 32, 0.f);
    fill_f32<<<grid1((size_t)n * 4),   B, 0, stream>>>(m1, (size_t)n * 4, -3.0e38f);
    fill_f32<<<grid1((size_t)n * 4),   B, 0, stream>>>(s1, (size_t)n * 4, 0.f);
    fill_f32<<<grid1((size_t)n),       B, 0, stream>>>(m2, (size_t)n, -3.0e38f);
    fill_f32<<<grid1((size_t)n),       B, 0, stream>>>(s2, (size_t)n, 0.f);
    fill_f32<<<1, B, 0, stream>>>(sums, 128, 0.f);
    fill_f32<<<1, B, 0, stream>>>(sumsq, 128, 0.f);

    // ---- layer 1: GEMM (WMMA bf16) + attention + segment softmax + scatter
    f32_to_bf16_k<<<grid1((size_t)n * 128), B, 0, stream>>>(x, x16, (size_t)n * 128);
    gemm_wmma_bf16<128><<<(n + 127) / 128, 256, 0, stream>>>(x16, W1, h1, n);
    attn_coeffs<<<grid1(n), B, 0, stream>>>(h1, a1s, a1d, als1, ald1, n, 4);
    edge_score_max_k<<<grid1(EE), B, 0, stream>>>(src, dst, E, n, als1, ald1, m1, eval, 4);
    edge_exp_sum_k  <<<grid1(EE), B, 0, stream>>>(dst, E, n, m1, s1, eval, 4);
    recip_k<<<grid1((size_t)n * 4), B, 0, stream>>>(s1, (size_t)n * 4);
    edge_aggregate_k<<<grid1((size_t)EE * 32), B, 0, stream>>>(
        src, dst, E, n, s1, eval, h1, agg1, 4);

    // ---- BatchNorm + ELU (fused, emits bf16 for GEMM2)
    bn_stats_k<<<2048, B, 0, stream>>>(agg1, n, sums, sumsq);
    bn_elu_bf16_k<<<grid1((size_t)n * 128), B, 0, stream>>>(agg1, sums, sumsq, gamma, beta, x16, n);

    // ---- layer 2 (H=1): GEMM (WMMA bf16) + attention + softmax + scatter
    gemm_wmma_bf16<32><<<(n + 127) / 128, 256, 0, stream>>>(x16, W2, h2, n);
    attn_coeffs<<<grid1(n), B, 0, stream>>>(h2, a2s, a2d, als2, ald2, n, 1);
    edge_score_max_k<<<grid1(EE), B, 0, stream>>>(src, dst, E, n, als2, ald2, m2, eval, 1);
    edge_exp_sum_k  <<<grid1(EE), B, 0, stream>>>(dst, E, n, m2, s2, eval, 1);
    recip_k<<<grid1(n), B, 0, stream>>>(s2, (size_t)n);
    edge_aggregate_k<<<grid1((size_t)EE * 32), B, 0, stream>>>(
        src, dst, E, n, s2, eval, h2, agg2, 1);

    // ---- classifier + log_softmax
    classify_k<<<grid1(n), B, 0, stream>>>(agg2, b2, Wc, bc, out, n);
    (void)n_in; (void)out_size; (void)ws_size;
}
